// SOLD2_71468255805975
// MI455X (gfx1250) — compile-verified, matrix operand order checked by reading, not run on hardware
//
#include <hip/hip_runtime.h>
#include <hip/hip_bf16.h>

typedef _Float16 v16h __attribute__((ext_vector_type(16)));
typedef _Float16 v8h  __attribute__((ext_vector_type(8)));
typedef float    v8f  __attribute__((ext_vector_type(8)));

#define NLINES 1200
#define PAD_S  8            // samples padded 5 -> 8 so 2 lines = one 16-row WMMA tile
#define NSAMP  5
#define NPTS   (NLINES*PAD_S)   // 9600 = 600 tiles of 16
#define DIMC   128
#define IMH    128
#define IMW    128
#define TILES  600
#define GAPP   0.1f
#define TOPKK  10

#define WM 2                 // 16-row tiles per wave (i side)
#define WN 4                 // 16-col tiles per wave (j side)
#define JOBS ((TILES/WM)*(TILES/WN))   // 300*150 = 45000 wave jobs

// ---------------------------------------------------------------------------
// Kernel 1: per-point bilinear descriptor sampling + L2 normalize -> f16 rows.
// One block (128 threads = 1 channel/thread) per padded sample point.
// ---------------------------------------------------------------------------
__global__ void __launch_bounds__(128)
sample_kernel(const float* __restrict__ seg1, const float* __restrict__ seg2,
              const float* __restrict__ desc1, const float* __restrict__ desc2,
              _Float16* __restrict__ d1, _Float16* __restrict__ d2,
              int* __restrict__ ns1, int* __restrict__ ns2)
{
    int p    = blockIdx.x;            // 0 .. 2*NPTS-1
    int side = p / NPTS;
    int idx  = p - side * NPTS;
    int line = idx >> 3;
    int k    = idx & 7;

    const float* seg = side ? seg2 : seg1;
    const float* dsc = side ? desc2 : desc1;
    _Float16*    out = side ? d2 : d1;
    int*       nsarr = side ? ns2 : ns1;

    float sy = seg[line*4+0], sx = seg[line*4+1];
    float ey = seg[line*4+2], ex = seg[line*4+3];
    float dy = ey - sy, dx = ex - sx;
    float len = sqrtf(dy*dy + dx*dx);
    float ns  = fminf(fmaxf(floorf(len * 0.125f), 2.0f), 5.0f);
    if (k == 0 && threadIdx.x == 0) nsarr[line] = (int)ns;
    bool valid = (float)k < ns;

    float inv = 1.0f / (ns - 1.0f);
    float py = sy + (float)k * dy * inv;
    float px = sx + (float)k * dx * inv;

    // grid_sample coordinate mapping (img = 512, feature map = 128)
    float xn  = 2.0f * px / 511.0f - 1.0f;
    float yn  = 2.0f * py / 511.0f - 1.0f;
    float ixf = ((xn + 1.0f) * 128.0f - 1.0f) * 0.5f;
    float iyf = ((yn + 1.0f) * 128.0f - 1.0f) * 0.5f;
    float x0f = floorf(ixf), y0f = floorf(iyf);
    float wx = ixf - x0f, wy = iyf - y0f;
    int x0 = (int)x0f, y0 = (int)y0f;

    int c = threadIdx.x;              // channel
    float v = 0.0f;
    if (valid) {
        const float* img = dsc + (size_t)c * (IMH * IMW);
        auto g = [&](int xi, int yi) -> float {
            bool inb = (xi >= 0) & (xi < IMW) & (yi >= 0) & (yi < IMH);
            int xc = xi < 0 ? 0 : (xi > IMW-1 ? IMW-1 : xi);
            int yc = yi < 0 ? 0 : (yi > IMH-1 ? IMH-1 : yi);
            float t = img[yc * IMW + xc];
            return inb ? t : 0.0f;
        };
        v = g(x0,  y0  ) * ((1.0f-wx)*(1.0f-wy))
          + g(x0+1,y0  ) * (wx*(1.0f-wy))
          + g(x0,  y0+1) * ((1.0f-wx)*wy)
          + g(x0+1,y0+1) * (wx*wy);
    }

    // block-wide sum of squares (4 waves of 32)
    float ss = v * v;
    #pragma unroll
    for (int m = 16; m >= 1; m >>= 1) ss += __shfl_xor(ss, m, 32);
    __shared__ float red[4];
    if ((threadIdx.x & 31) == 0) red[threadIdx.x >> 5] = ss;
    __syncthreads();
    float tot = red[0] + red[1] + red[2] + red[3];

    float o = (tot > 0.0f) ? v / sqrtf(tot) : 0.0f;
    out[(size_t)idx * DIMC + c] = (_Float16)o;
}

// ---------------------------------------------------------------------------
// Kernel 2: fused GEMM + block reduction -> line_scores (1200 x 1200).
// Register-blocked: one wave owns a WM x WN grid of 16x16 score tiles
// (32x64 sample points = 4x8 line pairs). K=128 -> 8 WMMAs per k-step,
// 32 v_wmma_f32_16x16x32_f16 per wave, operands fetched once per wave.
// C layout (ISA 7.12.2): lanes 0-15 hold rows 0-7 (even line), lanes 16-31
// rows 8-15 (odd line); lane%16 = column; 8-lane groups share one j-line.
// ---------------------------------------------------------------------------
__global__ void __launch_bounds__(256)
line_score_wmma_kernel(const _Float16* __restrict__ d1, const _Float16* __restrict__ d2,
                       const int* __restrict__ ns1, const int* __restrict__ ns2,
                       float* __restrict__ ls)
{
    int wave = threadIdx.x >> 5;
    int lane = threadIdx.x & 31;
    int job  = blockIdx.x * 8 + wave;          // 0 .. JOBS-1
    int jm   = job / (TILES / WN);             // 0 .. 299
    int jn   = job - jm * (TILES / WN);        // 0 .. 149
    int tm0  = jm * WM;
    int tn0  = jn * WN;
    int mrow = lane & 15;
    int half = lane >> 4;

    // A operand: lane holds row M=lane&15, K-chunks {k0+8h.., k0+16+8h..}
    const _Float16* abase = d1 + (size_t)(tm0 * 16 + mrow) * DIMC + half * 8;
    // B operand: symmetric layout, columns are d2 rows
    const _Float16* bbase = d2 + (size_t)(tn0 * 16 + mrow) * DIMC + half * 8;

    v8f acc[WM][WN];
    #pragma unroll
    for (int m = 0; m < WM; m++)
        #pragma unroll
        for (int n = 0; n < WN; n++)
            acc[m][n] = (v8f){};

    #pragma unroll
    for (int k0 = 0; k0 < DIMC; k0 += 32) {
        v16h a[WM], b[WN];
        #pragma unroll
        for (int m = 0; m < WM; m++) {
            const _Float16* pa = abase + (size_t)m * 16 * DIMC + k0;
            v8h lo = *(const v8h*)pa;
            v8h hi = *(const v8h*)(pa + 16);
            #pragma unroll
            for (int q = 0; q < 8; q++) { a[m][q] = lo[q]; a[m][8+q] = hi[q]; }
        }
        #pragma unroll
        for (int n = 0; n < WN; n++) {
            const _Float16* pb = bbase + (size_t)n * 16 * DIMC + k0;
            v8h lo = *(const v8h*)pb;
            v8h hi = *(const v8h*)(pb + 16);
            #pragma unroll
            for (int q = 0; q < 8; q++) { b[n][q] = lo[q]; b[n][8+q] = hi[q]; }
        }
        #pragma unroll
        for (int m = 0; m < WM; m++)
            #pragma unroll
            for (int n = 0; n < WN; n++)
                acc[m][n] = __builtin_amdgcn_wmma_f32_16x16x32_f16(
                                false, a[m], false, b[n], (short)0,
                                acc[m][n], false, false);
    }

    // Per-tile mask + mean/max reduction epilogue.
    int n4 = lane & 15;                       // column within a tile
    #pragma unroll
    for (int m = 0; m < WM; m++) {
        int li  = (tm0 + m) * 2 + half;       // i-line of this lane's rows
        int nsi = ns1[li];
        #pragma unroll
        for (int n = 0; n < WN; n++) {
            int jl  = (tn0 + n) * 2 + (n4 >> 3);
            int nsj = ns2[jl];
            bool colv = (n4 & 7) < nsj;
            v8f c = acc[m][n];

            // ls2: max over valid i-samples (per-lane over 8 VGPRs),
            // then mean over valid j-samples (xor-sum across 8-lane group).
            float vmax = -1.0f;
            #pragma unroll
            for (int r = 0; r < 8; r++) {
                float x = (r < nsi) ? c[r] : -1.0f;
                vmax = fmaxf(vmax, x);
            }
            float contrib = colv ? vmax : 0.0f;
            #pragma unroll
            for (int s = 1; s < 8; s <<= 1) contrib += __shfl_xor(contrib, s, 32);
            float ls2v = contrib / (float)nsj;

            // ls1: max over j-samples (xor-max across group, per row r),
            // then mean over valid i-samples.
            float rowsum = 0.0f;
            #pragma unroll
            for (int r = 0; r < 8; r++) {
                float x = colv ? c[r] : -1.0f;
                #pragma unroll
                for (int s = 1; s < 8; s <<= 1) x = fmaxf(x, __shfl_xor(x, s, 32));
                if (r < nsi) rowsum += x;
            }
            float ls1v = rowsum / (float)nsi;

            if ((lane & 7) == 0)
                ls[(size_t)li * NLINES + jl] = 0.5f * (ls1v + ls2v);
        }
    }
}

// ---------------------------------------------------------------------------
// Kernel 3: per row: top-10 (argsort-ascending tail semantics), recompute the
// 5x5 score blocks for the 10 lines (+ flipped), Needleman-Wunsch, argmax.
// One block (32 threads) per row. transposed=1 swaps the two sides.
// ---------------------------------------------------------------------------
__global__ void __launch_bounds__(32)
topk_nw_kernel(const float* __restrict__ ls,
               const _Float16* __restrict__ dA, const _Float16* __restrict__ dB,
               const int* __restrict__ nsA, const int* __restrict__ nsB,
               int transposed,
               float* __restrict__ nw_out, int* __restrict__ match_out)
{
    int i = blockIdx.x;
    __shared__ int   s_idx[TOPKK];
    __shared__ float s_nw[2 * TOPKK];

    if (threadIdx.x == 0) {
        float tv[TOPKK]; int ti[TOPKK];        // ascending by (value, index)
        #pragma unroll
        for (int q = 0; q < TOPKK; q++) { tv[q] = -1e30f; ti[q] = -1; }
        for (int j = 0; j < NLINES; j++) {
            float v = transposed ? ls[(size_t)j * NLINES + i]
                                 : ls[(size_t)i * NLINES + j];
            if (v > tv[0] || (v == tv[0] && j > ti[0])) {
                int p = 0;
                while (p < TOPKK - 1 &&
                       (tv[p+1] < v || (tv[p+1] == v && ti[p+1] < j))) {
                    tv[p] = tv[p+1]; ti[p] = ti[p+1]; p++;
                }
                tv[p] = v; ti[p] = j;
            }
        }
        #pragma unroll
        for (int q = 0; q < TOPKK; q++) s_idx[q] = ti[q];
    }
    __syncthreads();

    int t = threadIdx.x;
    if (t < 2 * TOPKK) {
        int q = t % TOPKK;
        bool flip = (t >= TOPKK);
        int j = s_idx[q];
        int nA = nsA[i], nB = nsB[j];

        float prev[NSAMP + 1];
        #pragma unroll
        for (int b = 0; b <= NSAMP; b++) prev[b] = 0.0f;
        for (int a = 0; a < NSAMP; a++) {
            float left = 0.0f;
            float newr[NSAMP + 1]; newr[0] = 0.0f;
            for (int b = 0; b < NSAMP; b++) {
                int bb = flip ? (NSAMP - 1 - b) : b;
                float sc;
                if (a < nA && bb < nB) {
                    const _Float16* ra = dA + (size_t)(i * PAD_S + a)  * DIMC;
                    const _Float16* rb = dB + (size_t)(j * PAD_S + bb) * DIMC;
                    float dot = 0.0f;
                    for (int cc = 0; cc < DIMC; cc++)
                        dot += (float)ra[cc] * (float)rb[cc];
                    sc = dot;
                } else {
                    sc = -1.0f;
                }
                sc -= GAPP;
                float cur = fmaxf(fmaxf(left, prev[b+1]), prev[b] + sc);
                left = cur; newr[b+1] = cur;
            }
            #pragma unroll
            for (int b = 0; b <= NSAMP; b++) prev[b] = newr[b];
        }
        s_nw[t] = prev[NSAMP];
        if (nw_out) nw_out[(size_t)i * (2 * TOPKK) + t] = prev[NSAMP];
    }
    __syncthreads();

    if (threadIdx.x == 0) {
        float best = s_nw[0]; int bt = 0;
        for (int t2 = 1; t2 < 2 * TOPKK; t2++)
            if (s_nw[t2] > best) { best = s_nw[t2]; bt = t2; }
        match_out[i] = s_idx[bt % TOPKK];
    }
}

// ---------------------------------------------------------------------------
// Kernel 4: mutual-consistency check, write matches as float into d_out[0..N).
// ---------------------------------------------------------------------------
__global__ void mutual_kernel(const int* __restrict__ m1, const int* __restrict__ m2,
                              float* __restrict__ out_matches)
{
    int i = blockIdx.x * blockDim.x + threadIdx.x;
    if (i < NLINES) {
        int m = m1[i];
        out_matches[i] = (m2[m] == i) ? (float)m : -1.0f;
    }
}

// ---------------------------------------------------------------------------
extern "C" void kernel_launch(void* const* d_in, const int* in_sizes, int n_in,
                              void* d_out, int out_size, void* d_ws, size_t ws_size,
                              hipStream_t stream)
{
    (void)in_sizes; (void)n_in; (void)out_size; (void)ws_size;
    const float* seg1  = (const float*)d_in[0];
    const float* seg2  = (const float*)d_in[1];
    const float* desc1 = (const float*)d_in[2];
    const float* desc2 = (const float*)d_in[3];
    float* out = (float*)d_out;           // [0,1200): matches, [1200,25200): nw

    char* ws = (char*)d_ws;
    _Float16* d1 = (_Float16*)ws;  ws += (size_t)NPTS * DIMC * sizeof(_Float16);
    _Float16* d2 = (_Float16*)ws;  ws += (size_t)NPTS * DIMC * sizeof(_Float16);
    int* ns1 = (int*)ws;           ws += NLINES * sizeof(int);
    int* ns2 = (int*)ws;           ws += NLINES * sizeof(int);
    float* lsbuf = (float*)ws;     ws += (size_t)NLINES * NLINES * sizeof(float);
    int* m1 = (int*)ws;            ws += NLINES * sizeof(int);
    int* m2 = (int*)ws;            ws += NLINES * sizeof(int);

    sample_kernel<<<2 * NPTS, 128, 0, stream>>>(seg1, seg2, desc1, desc2,
                                                d1, d2, ns1, ns2);
    line_score_wmma_kernel<<<JOBS / 8, 256, 0, stream>>>(d1, d2, ns1, ns2, lsbuf);
    topk_nw_kernel<<<NLINES, 32, 0, stream>>>(lsbuf, d1, d2, ns1, ns2, 0,
                                              out + NLINES, m1);
    topk_nw_kernel<<<NLINES, 32, 0, stream>>>(lsbuf, d2, d1, ns2, ns1, 1,
                                              nullptr, m2);
    mutual_kernel<<<(NLINES + 255) / 256, 256, 0, stream>>>(m1, m2, out);
}
